// MultiHeadAttn_61460982005937
// MI455X (gfx1250) — matrix operand order repaired and checked by reading, not compiled, not run on hardware
//
#include <hip/hip_runtime.h>

// ---------------- problem constants ----------------
#define B_   8
#define S_   2048
#define DM_  384
#define H_   2
#define DH_  64
#define IN_  128           // H_*DH_
#define MT_  (B_*S_)       // 16384 tokens

typedef _Float16 h16;
typedef __attribute__((ext_vector_type(8)))  _Float16 v8h;
typedef __attribute__((ext_vector_type(16))) _Float16 v16h;
typedef __attribute__((ext_vector_type(8)))  float    v8f;
typedef int vsi4 __attribute__((vector_size(16)));   // matches builtin's int4-vector param

#if __has_builtin(__builtin_amdgcn_global_load_async_to_lds_b128)
#define HAVE_ASYNC_B128 1
#else
#define HAVE_ASYNC_B128 0
#endif

static __device__ __forceinline__ v16h cat16(v8h lo, v8h hi) {
  return __builtin_shufflevector(lo, hi, 0,1,2,3,4,5,6,7,8,9,10,11,12,13,14,15);
}

static __device__ __forceinline__ v8f wmma_f16(v16h a, v16h b, v8f c) {
  // D = A(16x32 f16) * B(32x16 f16) + C(16x16 f32)
  return __builtin_amdgcn_wmma_f32_16x16x32_f16(false, a, false, b, (short)0, c, false, false);
}

// A fragment (16x32, f16). base = &M[row0][k0], row-major, leading dim ld.
// ISA layout: lanes 0-15 row m hold K 0-7 & 16-23; lanes 16-31 row m hold K 8-15 & 24-31.
static __device__ __forceinline__ v16h ldA(const h16* base, int ld, int lane) {
  int m = lane & 15, ko = (lane >> 4) * 8;
  const h16* p = base + (size_t)m * ld + ko;
  return cat16(*(const v8h*)p, *(const v8h*)(p + 16));
}

// B fragment (32x16, f16) from source stored as [n][k] row-major. base = &Mt[n0][k0].
// ISA layout: lane = column (n), lanes 0-15 hold K 0-7 & 16-23, lanes 16-31 hold K 8-15 & 24-31.
static __device__ __forceinline__ v16h ldB(const h16* base, int ld, int lane) {
  const h16* p = base + (size_t)(lane & 15) * ld + ((lane >> 4) << 3);
  return cat16(*(const v8h*)p, *(const v8h*)(p + 16));
}

// ---------------- kernel 1: x fp32 -> f16 ----------------
__global__ void k_cvt_x(const float* __restrict__ x, h16* __restrict__ Xh) {
  size_t i = ((size_t)blockIdx.x * blockDim.x + threadIdx.x) * 8;
  v8h o;
#pragma unroll
  for (int e = 0; e < 8; ++e) o[e] = (h16)x[i + e];
  *(v8h*)(Xh + i) = o;
}

// ---------------- kernel 2: transpose + convert weights ----------------
// qkv_w[k=384][n=384] -> WqkvT[n][k] f16 ; o_w[k=128][n=384] -> WoT[n][k] f16
__global__ void k_prep_w(const float* __restrict__ qkv_w, const float* __restrict__ o_w,
                         h16* __restrict__ WqkvT, h16* __restrict__ WoT) {
  int i = blockIdx.x * 256 + threadIdx.x;           // 0 .. 196607 exactly
  if (i < 384 * 384) {
    int n = i / 384, k = i % 384;
    WqkvT[n * 384 + k] = (h16)qkv_w[k * 384 + n];
  } else {
    int j = i - 384 * 384;
    int n = j / 128, k = j % 128;
    WoT[n * 128 + k] = (h16)o_w[k * 384 + n];
  }
}

// ---------------- kernel 3: QKV GEMM ----------------
// C[16384,384] = Xh @ WqkvT^T + bias; writes Q(scaled by 1/8)/K/V as f16 [b][h][s][d]
__global__ __launch_bounds__(256) void k_qkv(const h16* __restrict__ Xh,
                                             const h16* __restrict__ WqkvT,
                                             const float* __restrict__ qkv_b,
                                             h16* __restrict__ Qw, h16* __restrict__ Kw,
                                             h16* __restrict__ Vw) {
  __shared__ __align__(128) h16 Bs[128 * 32];
  int t = threadIdx.x, lane = t & 31, wave = t >> 5;
  int n0 = blockIdx.x * 128;
  int row0 = blockIdx.y * 128 + wave * 16;
  v8f acc[8] = {};

  for (int kt = 0; kt < 12; ++kt) {
    int k0 = kt * 32;
    __syncthreads();
    {
      int nr = t >> 1, ch = (t & 1) * 16;   // 128 rows x 32 halfs of WqkvT tile
      *(v16h*)&Bs[nr * 32 + ch] = *(const v16h*)(WqkvT + (size_t)(n0 + nr) * 384 + k0 + ch);
    }
    __syncthreads();
    v16h a = ldA(Xh + (size_t)row0 * DM_ + k0, DM_, lane);
#pragma unroll
    for (int j = 0; j < 8; ++j) {
      v16h bf = ldB(Bs + j * 16 * 32, 32, lane);
      acc[j] = wmma_f16(a, bf, acc[j]);
    }
  }

  // epilogue: bias, split into Q (x 1/sqrt(64)) / K / V head-major
#pragma unroll
  for (int j = 0; j < 8; ++j) {
    int n = n0 + j * 16 + (lane & 15);
    float bias = qkv_b[n];
    int which = n >> 7;           // 0:Q 1:K 2:V
    int nh = n & 127, hh = nh >> 6, d = nh & 63;
    h16* dst = (which == 0) ? Qw : (which == 1 ? Kw : Vw);
    float scl = (which == 0) ? 0.125f : 1.0f;
    int r0 = row0 + ((lane >> 4) << 3);
#pragma unroll
    for (int r = 0; r < 8; ++r) {
      int row = r0 + r, bb = row >> 11, s = row & (S_ - 1);
      dst[((size_t)(bb * H_ + hh) * S_ + s) * DH_ + d] = (h16)((acc[j][r] + bias) * scl);
    }
  }
}

// ---------------- kernel 4: flash attention ----------------
// per block: one (b,h), 128 query rows (8 waves x 16). Online softmax, AV f16 out.
__global__ __launch_bounds__(256) void k_attn(const h16* __restrict__ Qw,
                                              const h16* __restrict__ Kw,
                                              const h16* __restrict__ Vw,
                                              const unsigned char* __restrict__ msk,
                                              h16* __restrict__ AV) {
  __shared__ __align__(128) h16 Kt[64 * 64];        // [key][d]
  __shared__ __align__(128) h16 Vt[64 * 64];        // [d][key] (transposed)
  __shared__ __align__(128) h16 Pb[8 * 16 * 64];    // per-wave P bounce buffer
  int t = threadIdx.x, lane = t & 31, wave = t >> 5;
  int bb = blockIdx.z, hh = blockIdx.y, q0 = blockIdx.x * 128;
  const size_t bh = (size_t)(bb * H_ + hh) * S_;
  int qr0 = q0 + wave * 16;

  v16h qa0 = ldA(Qw + (bh + qr0) * DH_ + 0, DH_, lane);
  v16h qa1 = ldA(Qw + (bh + qr0) * DH_ + 32, DH_, lane);
  v8f o[4] = {};
  float mst[8], lst[8];
#pragma unroll
  for (int r = 0; r < 8; ++r) { mst[r] = -3.0e38f; lst[r] = 0.f; }

  int krow = t >> 2, kch = (t & 3) * 16;
  const unsigned char* mrow = msk + (size_t)bb * S_;

  for (int s0 = 0; s0 < S_; s0 += 64) {
    __syncthreads();
    // stage K tile [64 keys][64 d] into LDS (async DMA path on CDNA5)
    const h16* gk = Kw + (bh + s0 + krow) * DH_ + kch;
    h16* lk = &Kt[krow * 64 + kch];
#if HAVE_ASYNC_B128
    __builtin_amdgcn_global_load_async_to_lds_b128(
        (__attribute__((address_space(1))) vsi4*)gk,
        (__attribute__((address_space(3))) vsi4*)lk, 0, 0);
    __builtin_amdgcn_global_load_async_to_lds_b128(
        (__attribute__((address_space(1))) vsi4*)(gk + 8),
        (__attribute__((address_space(3))) vsi4*)(lk + 8), 0, 0);
#else
    *(v16h*)lk = *(const v16h*)gk;
#endif
    // stage V^T tile [64 d][64 keys]
    const h16* gv = Vw + (bh + s0 + krow) * DH_ + kch;
    v8h va = *(const v8h*)gv, vb = *(const v8h*)(gv + 8);
#pragma unroll
    for (int e = 0; e < 8; ++e) {
      Vt[(kch + e) * 64 + krow] = va[e];
      Vt[(kch + 8 + e) * 64 + krow] = vb[e];
    }
#if HAVE_ASYNC_B128
#if __has_builtin(__builtin_amdgcn_s_wait_asynccnt)
    __builtin_amdgcn_s_wait_asynccnt(0);
#else
    asm volatile("s_wait_asynccnt 0x0" ::: "memory");
#endif
#endif
    __syncthreads();

    // scores = Q @ K^T (k-dim = d = 64)
    v8f sc[4] = {};
#pragma unroll
    for (int j = 0; j < 4; ++j) {
      v16h b0 = ldB(Kt + (j * 16) * 64 + 0, 64, lane);
      v16h b1 = ldB(Kt + (j * 16) * 64 + 32, 64, lane);
      sc[j] = wmma_f16(qa0, b0, sc[j]);
      sc[j] = wmma_f16(qa1, b1, sc[j]);
    }
    // key mask
#pragma unroll
    for (int j = 0; j < 4; ++j) {
      if (!mrow[s0 + j * 16 + (lane & 15)]) {
#pragma unroll
        for (int r = 0; r < 8; ++r) sc[j][r] = -3.0e38f;
      }
    }
    // online softmax: row max / rescale / row sum (16-lane xor reductions, wave32)
    float tmax[8];
#pragma unroll
    for (int r = 0; r < 8; ++r)
      tmax[r] = fmaxf(fmaxf(sc[0][r], sc[1][r]), fmaxf(sc[2][r], sc[3][r]));
#pragma unroll
    for (int off = 1; off < 16; off <<= 1)
#pragma unroll
      for (int r = 0; r < 8; ++r) tmax[r] = fmaxf(tmax[r], __shfl_xor(tmax[r], off, 32));

    float al[8], rsum[8];
#pragma unroll
    for (int r = 0; r < 8; ++r) {
      float nm = fmaxf(mst[r], tmax[r]);
      al[r] = __expf(mst[r] - nm);
      mst[r] = nm;
      float rs = 0.f;
#pragma unroll
      for (int j = 0; j < 4; ++j) { float p = __expf(sc[j][r] - nm); sc[j][r] = p; rs += p; }
      rsum[r] = rs;
    }
#pragma unroll
    for (int off = 1; off < 16; off <<= 1)
#pragma unroll
      for (int r = 0; r < 8; ++r) rsum[r] += __shfl_xor(rsum[r], off, 32);
#pragma unroll
    for (int r = 0; r < 8; ++r) lst[r] = lst[r] * al[r] + rsum[r];
#pragma unroll
    for (int j = 0; j < 4; ++j)
#pragma unroll
      for (int r = 0; r < 8; ++r) o[j][r] *= al[r];

    // bounce P (C-layout) through LDS to A-layout
    h16* pb = Pb + wave * 16 * 64;
    int mb = (lane >> 4) << 3;
#pragma unroll
    for (int j = 0; j < 4; ++j)
#pragma unroll
      for (int r = 0; r < 8; ++r)
        pb[(mb + r) * 64 + j * 16 + (lane & 15)] = (h16)sc[j][r];
    __syncthreads();
    v16h pa0 = ldA(pb + 0, 64, lane);
    v16h pa1 = ldA(pb + 32, 64, lane);
    // O += P @ V (k-dim = keys = 64)
#pragma unroll
    for (int j = 0; j < 4; ++j) {
      v16h b0 = ldB(Vt + (j * 16) * 64 + 0, 64, lane);
      v16h b1 = ldB(Vt + (j * 16) * 64 + 32, 64, lane);
      o[j] = wmma_f16(pa0, b0, o[j]);
      o[j] = wmma_f16(pa1, b1, o[j]);
    }
  }

#pragma unroll
  for (int r = 0; r < 8; ++r) lst[r] = 1.0f / lst[r];
  int sb = qr0 + ((lane >> 4) << 3);
#pragma unroll
  for (int j = 0; j < 4; ++j)
#pragma unroll
    for (int r = 0; r < 8; ++r)
      AV[((size_t)(bb * S_ + sb + r)) * IN_ + hh * DH_ + j * 16 + (lane & 15)] =
          (h16)(o[j][r] * lst[r]);
}

// ---------------- kernel 5: output GEMM + residual + LayerNorm ----------------
__global__ __launch_bounds__(256) void k_out(const h16* __restrict__ AV,
                                             const h16* __restrict__ WoT,
                                             const float* __restrict__ x,
                                             const float* __restrict__ g,
                                             const float* __restrict__ bt,
                                             float* __restrict__ out) {
  __shared__ float red[2][16];
  int t = threadIdx.x, lane = t & 31, wave = t >> 5;
  int row0 = blockIdx.x * 16;
  if (t < 16) { red[0][t] = 0.f; red[1][t] = 0.f; }
  __syncthreads();
  int n0 = wave * 48;                 // 8 waves x 48 cols = 384
  v8f acc[3] = {};
#pragma unroll
  for (int kt = 0; kt < 4; ++kt) {
    int k0 = kt * 32;
    v16h a = ldA(AV + (size_t)row0 * IN_ + k0, IN_, lane);
#pragma unroll
    for (int j = 0; j < 3; ++j) {
      v16h bf = ldB(WoT + (size_t)(n0 + j * 16) * IN_ + k0, IN_, lane);
      acc[j] = wmma_f16(a, bf, acc[j]);
    }
  }
  float y[3][8], ps[8] = {}, pq[8] = {};
  int rl0 = (lane >> 4) << 3;
#pragma unroll
  for (int j = 0; j < 3; ++j) {
    int col = n0 + j * 16 + (lane & 15);
#pragma unroll
    for (int r = 0; r < 8; ++r) {
      float v = acc[j][r] + x[(size_t)(row0 + rl0 + r) * DM_ + col];
      y[j][r] = v; ps[r] += v; pq[r] += v * v;
    }
  }
#pragma unroll
  for (int off = 1; off < 16; off <<= 1)
#pragma unroll
    for (int r = 0; r < 8; ++r) { ps[r] += __shfl_xor(ps[r], off, 32); pq[r] += __shfl_xor(pq[r], off, 32); }
  if ((lane & 15) == 0) {
#pragma unroll
    for (int r = 0; r < 8; ++r) { atomicAdd(&red[0][rl0 + r], ps[r]); atomicAdd(&red[1][rl0 + r], pq[r]); }
  }
  __syncthreads();
#pragma unroll
  for (int r = 0; r < 8; ++r) {
    float mu = red[0][rl0 + r] * (1.0f / DM_);
    float var = red[1][rl0 + r] * (1.0f / DM_) - mu * mu;
    float rstd = rsqrtf(var + 1e-3f);
    int row = row0 + rl0 + r;
#pragma unroll
    for (int j = 0; j < 3; ++j) {
      int col = n0 + j * 16 + (lane & 15);
      out[(size_t)row * DM_ + col] = (y[j][r] - mu) * rstd * g[col] + bt[col];
    }
  }
}

// ---------------- host launcher ----------------
extern "C" void kernel_launch(void* const* d_in, const int* in_sizes, int n_in,
                              void* d_out, int out_size, void* d_ws, size_t ws_size,
                              hipStream_t stream) {
  (void)in_sizes; (void)n_in; (void)out_size; (void)ws_size;
  const float* x            = (const float*)d_in[0];
  const unsigned char* mask = (const unsigned char*)d_in[1];
  const float* qkv_w        = (const float*)d_in[2];
  const float* qkv_b        = (const float*)d_in[3];
  const float* o_w          = (const float*)d_in[4];
  const float* ln_g         = (const float*)d_in[5];
  const float* ln_b         = (const float*)d_in[6];
  float* out = (float*)d_out;

  char* ws = (char*)d_ws;
  size_t off = 0;
  auto take = [&](size_t bytes) { void* p = ws + off; off += (bytes + 255) & ~(size_t)255; return p; };
  h16* Xh    = (h16*)take((size_t)MT_ * DM_ * sizeof(h16));   // 12 MB
  h16* WqkvT = (h16*)take((size_t)384 * 384 * sizeof(h16));
  h16* WoT   = (h16*)take((size_t)384 * 128 * sizeof(h16));
  h16* Qw    = (h16*)take((size_t)MT_ * IN_ * sizeof(h16));   // 4 MB
  h16* Kw    = (h16*)take((size_t)MT_ * IN_ * sizeof(h16));
  h16* Vw    = (h16*)take((size_t)MT_ * IN_ * sizeof(h16));
  h16* AV    = (h16*)take((size_t)MT_ * IN_ * sizeof(h16));

  k_cvt_x<<<(MT_ * DM_ / 8) / 256, 256, 0, stream>>>(x, Xh);
  k_prep_w<<<(384 * 384 + 384 * 128) / 256, 256, 0, stream>>>(qkv_w, o_w, WqkvT, WoT);
  k_qkv<<<dim3(3, MT_ / 128), 256, 0, stream>>>(Xh, WqkvT, qkv_b, Qw, Kw, Vw);
  k_attn<<<dim3(S_ / 128, H_, B_), 256, 0, stream>>>(Qw, Kw, Vw, mask, AV);
  k_out<<<MT_ / 16, 256, 0, stream>>>(AV, WoT, x, ln_g, ln_b, out);
}
